// MambaPatchMOL_87986700025948
// MI455X (gfx1250) — matrix operand order repaired and checked by reading, not compiled, not run on hardware
//
#include <hip/hip_runtime.h>
#include <math.h>

// ---------- model constants ----------
#define NSEQ   64          // B * h1 * h2 = 4*4*4
#define TLEN   128
#define ROWS   (NSEQ*TLEN) // 8192 tokens
#define DM     768
#define DI     1536
#define DSTATE 16
#define DTRANK 48
#define DTPAD  64          // DT_RANK padded to multiple of 32 for WMMA K loop
#define XDW    (DTRANK + 2*DSTATE)  // 80
#define DCONV  4

typedef __attribute__((ext_vector_type(16))) __bf16 v16bf;
typedef __attribute__((ext_vector_type(8)))  __bf16 v8bf;
typedef __attribute__((ext_vector_type(8)))  float  v8f;

// =====================================================================
// WMMA bf16 GEMM:  C[M][N] (f32) = A[M][K] (bf16) * W[N][K]^T (bf16)
// Register-blocked: each wave owns a (16*MT) x (16*NT) output block.
// Per 32-wide k-step: MT A-fragments + NT B-fragments feed MT*NT WMMAs
// (1.5 b128 loads per WMMA at MT=2,NT=4).
// Optional accumulate into existing C, optional bias[N].
// K mult of 32; M/16 mult of MT; N/16 mult of NT.
// =====================================================================
template<int MT, int NT>
__global__ __launch_bounds__(256) void gemm_bf16_blk(
    const __bf16* __restrict__ A, const __bf16* __restrict__ W,
    float* __restrict__ C, const float* __restrict__ bias,
    int M, int N, int K, int acc)
{
    const int lane = threadIdx.x & 31;
    const int wave = threadIdx.x >> 5;
    const int grp  = blockIdx.x * 8 + wave;
    const int ngrp = (N >> 4) / NT;
    const int tot  = ((M >> 4) / MT) * ngrp;
    if (grp >= tot) return;                  // wave-uniform: EXEC stays all-ones
    const int mg = grp / ngrp;
    const int ng = grp - mg * ngrp;

    const int lhi = lane >> 4;               // 0 or 1
    const int llo = lane & 15;
    const int m0  = (mg * MT) << 4;          // first output row of the block
    const int n0  = (ng * NT) << 4;          // first output column of the block

    v8f c[MT][NT];
    if (acc) {
#pragma unroll
        for (int i = 0; i < MT; ++i)
#pragma unroll
            for (int j = 0; j < NT; ++j)
#pragma unroll
                for (int r = 0; r < 8; ++r)
                    c[i][j][r] = C[(size_t)(m0 + (i << 4) + (lhi << 3) + r) * N
                                   + n0 + (j << 4) + llo];
    } else {
#pragma unroll
        for (int i = 0; i < MT; ++i)
#pragma unroll
            for (int j = 0; j < NT; ++j)
#pragma unroll
                for (int r = 0; r < 8; ++r) c[i][j][r] = 0.0f;
    }

    const __bf16* __restrict__ arow[MT];
#pragma unroll
    for (int i = 0; i < MT; ++i)
        arow[i] = A + (size_t)(m0 + (i << 4) + llo) * K;   // A row (M = lane%16)
    const __bf16* __restrict__ brow[NT];
#pragma unroll
    for (int j = 0; j < NT; ++j)
        brow[j] = W + (size_t)(n0 + (j << 4) + llo) * K + (lhi << 4);

    for (int k0 = 0; k0 < K; k0 += 32) {
        // A 16x32 bf16 layout: lanes<16 -> K = k0 + {0..15}
        //                      lanes>=16 -> K = k0 + {16..31}
        const int ka = k0 + (lhi << 4);
        v16bf a[MT];
#pragma unroll
        for (int i = 0; i < MT; ++i) {
            v8bf alo = *(const v8bf*)(arow[i] + ka);
            v8bf ahi = *(const v8bf*)(arow[i] + ka + 8);
#pragma unroll
            for (int e = 0; e < 8; ++e) { a[i][e] = alo[e]; a[i][e + 8] = ahi[e]; }
            if (k0 + 32 < K)                 // WGP-scope prefetch of next A k-block
                __builtin_prefetch(arow[i] + ka + 32, 0, 1);
        }
#pragma unroll
        for (int j = 0; j < NT; ++j) {
            // B 32x16 bf16: lane n holds column n; lanes<16 K=k0..15, lanes>=16 K=k0+16..31
            v16bf b = *(const v16bf*)(brow[j] + k0);
#pragma unroll
            for (int i = 0; i < MT; ++i)
                c[i][j] = __builtin_amdgcn_wmma_f32_16x16x32_bf16(
                    false, a[i], false, b, (short)0, c[i][j], false, false);
        }
    }

#pragma unroll
    for (int i = 0; i < MT; ++i) {
        const int mrow = m0 + (i << 4) + (lhi << 3);   // D rows: r + 8*(lane>>4)
#pragma unroll
        for (int j = 0; j < NT; ++j) {
            const int nb = n0 + (j << 4) + llo;
            const float bv = bias ? bias[nb] : 0.0f;
#pragma unroll
            for (int r = 0; r < 8; ++r)
                C[(size_t)(mrow + r) * N + nb] = c[i][j][r] + bv;
        }
    }
}

// ---------------- f32 -> bf16 convert ----------------
__global__ __launch_bounds__(256) void wcvt_kernel(const float* __restrict__ in,
                                                   __bf16* __restrict__ out, int n)
{
    int i = blockIdx.x * 256 + threadIdx.x;
    if (i < n) out[i] = (__bf16)in[i];
}

// dt_proj_w [rows][48] -> bf16 [rows][64] zero padded
__global__ __launch_bounds__(256) void wcvt_pad_kernel(const float* __restrict__ in,
                                                       __bf16* __restrict__ out, int rows)
{
    int i = blockIdx.x * 256 + threadIdx.x;
    if (i >= rows * DTPAD) return;
    int r = i / DTPAD, cix = i - r * DTPAD;
    out[i] = (cix < DTRANK) ? (__bf16)in[r * DTRANK + cix] : (__bf16)0.0f;
}

// dt_low slice of xdbl [ROWS][80] -> bf16 [ROWS][64] zero padded
__global__ __launch_bounds__(256) void stage_dt_kernel(const float* __restrict__ xdbl,
                                                       __bf16* __restrict__ out)
{
    int i = blockIdx.x * 256 + threadIdx.x;
    if (i >= ROWS * DTPAD) return;
    int r = i / DTPAD, cix = i - r * DTPAD;
    out[i] = (cix < DTRANK) ? (__bf16)xdbl[(size_t)r * XDW + cix] : (__bf16)0.0f;
}

// ---------------- patchify: x(4,128,3,64,64) -> bf16 [8192][768] ----------------
__global__ __launch_bounds__(256) void patchify_kernel(const float* __restrict__ x,
                                                       __bf16* __restrict__ out)
{
    int idx = blockIdx.x * 256 + threadIdx.x;
    if (idx >= ROWS * DM) return;
    int row = idx / DM, feat = idx - row * DM;
    int t = row & (TLEN - 1), seq = row / TLEN;
    int b = seq >> 4, rem = seq & 15, pi = rem >> 2, pj = rem & 3;
    int f = feat >> 8, pr = (feat >> 4) & 15, pc = feat & 15;
    size_t src = ((((size_t)b * TLEN + t) * 3 + f) * 64 + (pi * 16 + pr)) * 64
               + (pj * 16 + pc);
    out[idx] = (__bf16)x[src];
}

// ---------------- unpatchify: [8192][768] f32 -> d_out(4,128,3,64,64) ----------------
__global__ __launch_bounds__(256) void unpatchify_kernel(const float* __restrict__ y,
                                                         float* __restrict__ out)
{
    int idx = blockIdx.x * 256 + threadIdx.x;
    if (idx >= ROWS * DM) return;
    int ww = idx & 63;
    int hh = (idx >> 6) & 63;
    int f  = (idx >> 12) % 3;
    int t  = (idx / 12288) & (TLEN - 1);
    int b  = idx / 1572864;
    int pi = hh >> 4, pr = hh & 15, pj = ww >> 4, pc = ww & 15;
    int seq = b * 16 + pi * 4 + pj;
    int row = seq * TLEN + t;
    int feat = f * 256 + pr * 16 + pc;
    out[idx] = y[(size_t)row * DM + feat];
}

// ---------------- layernorm over DM, output bf16 ----------------
__global__ __launch_bounds__(256) void layernorm_kernel(const float* __restrict__ x,
                                                        const float* __restrict__ w,
                                                        const float* __restrict__ b,
                                                        __bf16* __restrict__ out)
{
    __shared__ float ssum[256], ssq[256];
    const int tid = threadIdx.x;
    const int row = blockIdx.x;
    const float* xr = x + (size_t)row * DM;
    float v0 = xr[tid], v1 = xr[tid + 256], v2 = xr[tid + 512];
    ssum[tid] = v0 + v1 + v2;
    ssq[tid]  = v0 * v0 + v1 * v1 + v2 * v2;
    __syncthreads();
    for (int off = 128; off > 0; off >>= 1) {
        if (tid < off) { ssum[tid] += ssum[tid + off]; ssq[tid] += ssq[tid + off]; }
        __syncthreads();
    }
    float mu  = ssum[0] * (1.0f / DM);
    float var = ssq[0] * (1.0f / DM) - mu * mu;
    float inv = rsqrtf(var + 1e-5f);
    __bf16* o = out + (size_t)row * DM;
    o[tid]       = (__bf16)((v0 - mu) * inv * w[tid]       + b[tid]);
    o[tid + 256] = (__bf16)((v1 - mu) * inv * w[tid + 256] + b[tid + 256]);
    o[tid + 512] = (__bf16)((v2 - mu) * inv * w[tid + 512] + b[tid + 512]);
}

// ---------------- causal depthwise conv (D_CONV=4) + bias + SiLU ----------------
__global__ __launch_bounds__(256) void conv_silu_kernel(const float* __restrict__ up,
                                                        const float* __restrict__ cw,
                                                        const float* __restrict__ cb,
                                                        float* __restrict__ u,
                                                        __bf16* __restrict__ ubf)
{
    size_t idx = (size_t)blockIdx.x * 256 + threadIdx.x;
    if (idx >= (size_t)ROWS * DI) return;
    int d = idx % DI;
    int t = (idx / DI) & (TLEN - 1);
    int s = idx / ((size_t)DI * TLEN);
    float acc = cb[d];
#pragma unroll
    for (int k = 0; k < DCONV; ++k) {
        int tt = t + k - (DCONV - 1);
        if (tt >= 0)
            acc += cw[d * DCONV + k] * up[((size_t)(s * TLEN + tt)) * DI + d];
    }
    float val = acc / (1.0f + __expf(-acc));   // silu
    u[idx]   = val;
    ubf[idx] = (__bf16)val;
}

// ---------------- softplus(x + dt_bias[d]) in place ----------------
__global__ __launch_bounds__(256) void softplus_kernel(float* __restrict__ buf,
                                                       const float* __restrict__ dpb)
{
    size_t idx = (size_t)blockIdx.x * 256 + threadIdx.x;
    if (idx >= (size_t)ROWS * DI) return;
    int d = idx % DI;
    float xv = buf[idx] + dpb[d];
    buf[idx] = (xv > 20.0f) ? xv : log1pf(__expf(xv));
}

// ---------------- selective scan, fused with D_skip + silu(z) gate ----------------
__global__ __launch_bounds__(256) void scan_kernel(const float* __restrict__ delta,
                                                   const float* __restrict__ u,
                                                   const float* __restrict__ z,
                                                   const float* __restrict__ xdbl,
                                                   const float* __restrict__ Alog,
                                                   const float* __restrict__ Dsk,
                                                   __bf16* __restrict__ ybf)
{
    int id = blockIdx.x * 256 + threadIdx.x;   // NSEQ*DI = 98304 threads exactly
    int d = id % DI;
    int s = id / DI;
    float A[DSTATE], h[DSTATE];
#pragma unroll
    for (int n = 0; n < DSTATE; ++n) { A[n] = -__expf(Alog[d * DSTATE + n]); h[n] = 0.0f; }
    const float Dv = Dsk[d];
    for (int t = 0; t < TLEN; ++t) {
        size_t base = (size_t)(s * TLEN + t);
        float dlt = delta[base * DI + d];
        float uv  = u[base * DI + d];
        float du  = dlt * uv;
        const float* bc = xdbl + base * XDW;
        float y = 0.0f;
#pragma unroll
        for (int n = 0; n < DSTATE; ++n) {
            h[n] = __expf(dlt * A[n]) * h[n] + du * bc[DTRANK + n];
            y += h[n] * bc[DTRANK + DSTATE + n];
        }
        y += uv * Dv;
        float zv = z[base * DI + d];
        y *= zv / (1.0f + __expf(-zv));
        ybf[base * DI + d] = (__bf16)y;
    }
}

// =====================================================================
static inline void launch_gemm(const __bf16* A, const __bf16* W, float* C,
                               const float* bias, int M, int N, int K, int acc,
                               hipStream_t s)
{
    const int ntn = N / 16;          // M is always ROWS here -> M/32 is exact
    if ((ntn & 3) == 0) {
        int groups = (M / 32) * (ntn / 4);
        hipLaunchKernelGGL((gemm_bf16_blk<2, 4>), dim3((groups + 7) / 8), dim3(256), 0, s,
                           A, W, C, bias, M, N, K, acc);
    } else {
        int groups = (M / 32) * ntn;
        hipLaunchKernelGGL((gemm_bf16_blk<2, 1>), dim3((groups + 7) / 8), dim3(256), 0, s,
                           A, W, C, bias, M, N, K, acc);
    }
}

static inline int ceil_div(size_t a, int b) { return (int)((a + b - 1) / b); }

extern "C" void kernel_launch(void* const* d_in, const int* in_sizes, int n_in,
                              void* d_out, int out_size, void* d_ws, size_t ws_size,
                              hipStream_t stream)
{
    (void)in_sizes; (void)n_in; (void)out_size; (void)ws_size;
    const float* x          = (const float*)d_in[0];
    const float* lin_in_w   = (const float*)d_in[1];
    const float* lin_in_b   = (const float*)d_in[2];
    const float* lin_out_w  = (const float*)d_in[3];
    const float* lin_out_b  = (const float*)d_in[4];
    const float* norm_w     = (const float*)d_in[5];
    const float* norm_b     = (const float*)d_in[6];
    const float* in_proj_w  = (const float*)d_in[7];
    const float* conv_w     = (const float*)d_in[8];
    const float* conv_b     = (const float*)d_in[9];
    const float* x_proj_w   = (const float*)d_in[10];
    const float* dt_proj_w  = (const float*)d_in[11];
    const float* dt_proj_b  = (const float*)d_in[12];
    const float* A_log      = (const float*)d_in[13];
    const float* D_skip     = (const float*)d_in[14];
    const float* out_proj_w = (const float*)d_in[15];
    float* out = (float*)d_out;

    // -------- workspace layout (all offsets 256B aligned) --------
    char* ws = (char*)d_ws;
    const size_t SZ_H   = (size_t)ROWS * DM * 4;        // 25.2 MB  f32 residual
    const size_t SZ_STG = (size_t)ROWS * DI * 2;        // 25.2 MB  bf16 A staging
    const size_t SZ_WBF = (size_t)2 * DI * DM * 2;      //  4.7 MB  bf16 weights
    const size_t SZ_DI  = (size_t)ROWS * DI * 4;        // 50.3 MB  f32
    float*  hres = (float*)(ws);
    __bf16* stg  = (__bf16*)(ws + SZ_H);
    __bf16* wbf  = (__bf16*)(ws + SZ_H + SZ_STG);
    float*  upre = (float*)(ws + SZ_H + SZ_STG + SZ_WBF);          // also delta, also final y
    float*  zbuf = (float*)(ws + SZ_H + SZ_STG + SZ_WBF + SZ_DI);
    float*  ubuf = (float*)(ws + SZ_H + SZ_STG + SZ_WBF + 2 * SZ_DI);
    float*  xdbl = (float*)(ws + SZ_H + SZ_STG + SZ_WBF + 3 * SZ_DI);

    const size_t nTok768  = (size_t)ROWS * DM;
    const size_t nTok1536 = (size_t)ROWS * DI;

    // ---- patch embed: h = patchify(x) @ lin_in_w^T + b ----
    hipLaunchKernelGGL(patchify_kernel, dim3(ceil_div(nTok768, 256)), dim3(256), 0, stream, x, stg);
    hipLaunchKernelGGL(wcvt_kernel, dim3(ceil_div((size_t)DM * DM, 256)), dim3(256), 0, stream,
                       lin_in_w, wbf, DM * DM);
    launch_gemm(stg, wbf, hres, lin_in_b, ROWS, DM, DM, 0, stream);

    for (int layer = 0; layer < 2; ++layer) {
        const float* nw  = norm_w + layer * DM;
        const float* nb  = norm_b + layer * DM;
        const float* ipw = in_proj_w + (size_t)layer * 2 * DI * DM;
        const float* cw  = conv_w + (size_t)layer * DI * DCONV;
        const float* cb  = conv_b + (size_t)layer * DI;
        const float* xpw = x_proj_w + (size_t)layer * XDW * DI;
        const float* dpw = dt_proj_w + (size_t)layer * DI * DTRANK;
        const float* dpb = dt_proj_b + (size_t)layer * DI;
        const float* Al  = A_log + (size_t)layer * DI * DSTATE;
        const float* Dsk = D_skip + (size_t)layer * DI;
        const float* opw = out_proj_w + (size_t)layer * DM * DI;

        // layernorm -> bf16 staging
        hipLaunchKernelGGL(layernorm_kernel, dim3(ROWS), dim3(256), 0, stream, hres, nw, nb, stg);

        // in_proj: u_pre = hn @ W[0:DI]^T ; z = hn @ W[DI:2DI]^T
        hipLaunchKernelGGL(wcvt_kernel, dim3(ceil_div((size_t)2 * DI * DM, 256)), dim3(256), 0,
                           stream, ipw, wbf, 2 * DI * DM);
        launch_gemm(stg, wbf, upre, nullptr, ROWS, DI, DM, 0, stream);
        launch_gemm(stg, wbf + (size_t)DI * DM, zbuf, nullptr, ROWS, DI, DM, 0, stream);

        // causal conv + SiLU -> u (f32) and u_bf (staging)
        hipLaunchKernelGGL(conv_silu_kernel, dim3(ceil_div(nTok1536, 256)), dim3(256), 0, stream,
                           upre, cw, cb, ubuf, stg);

        // x_proj: xdbl = u_bf @ x_proj_w^T  (N=80, K=1536)
        hipLaunchKernelGGL(wcvt_kernel, dim3(ceil_div((size_t)XDW * DI, 256)), dim3(256), 0,
                           stream, xpw, wbf, XDW * DI);
        launch_gemm(stg, wbf, xdbl, nullptr, ROWS, XDW, DI, 0, stream);

        // dt: stage dt_low padded, GEMM (K=64), then softplus(+bias)
        hipLaunchKernelGGL(stage_dt_kernel, dim3(ceil_div((size_t)ROWS * DTPAD, 256)), dim3(256),
                           0, stream, xdbl, stg);
        hipLaunchKernelGGL(wcvt_pad_kernel, dim3(ceil_div((size_t)DI * DTPAD, 256)), dim3(256),
                           0, stream, dpw, wbf, DI);
        launch_gemm(stg, wbf, upre, nullptr, ROWS, DI, DTPAD, 0, stream);   // upre now = delta_pre
        hipLaunchKernelGGL(softplus_kernel, dim3(ceil_div(nTok1536, 256)), dim3(256), 0, stream,
                           upre, dpb);

        // selective scan fused with D_skip and silu(z) gate -> y_bf (staging)
        hipLaunchKernelGGL(scan_kernel, dim3((NSEQ * DI) / 256), dim3(256), 0, stream,
                           upre, ubuf, zbuf, xdbl, Al, Dsk, stg);

        // out_proj with residual accumulate: h += y_bf @ out_proj_w^T
        hipLaunchKernelGGL(wcvt_kernel, dim3(ceil_div((size_t)DM * DI, 256)), dim3(256), 0,
                           stream, opw, wbf, DM * DI);
        launch_gemm(stg, wbf, hres, nullptr, ROWS, DM, DI, 1, stream);
    }

    // ---- head: y = h @ lin_out_w^T + b, then unpatchify ----
    hipLaunchKernelGGL(wcvt_kernel, dim3(ceil_div(nTok768, 256)), dim3(256), 0, stream,
                       hres, stg, ROWS * DM);
    hipLaunchKernelGGL(wcvt_kernel, dim3(ceil_div((size_t)DM * DM, 256)), dim3(256), 0, stream,
                       lin_out_w, wbf, DM * DM);
    launch_gemm(stg, wbf, upre, lin_out_b, ROWS, DM, DM, 0, stream);        // upre = final y
    hipLaunchKernelGGL(unpatchify_kernel, dim3(ceil_div(nTok768, 256)), dim3(256), 0, stream,
                       upre, out);
}
